// model4_14877766713701
// MI455X (gfx1250) — matrix operand order repaired and checked by reading, not compiled, not run on hardware
//
#include <hip/hip_runtime.h>

typedef __attribute__((ext_vector_type(16))) __bf16 v16bf;
typedef __attribute__((ext_vector_type(8)))  float  v8f;

#define SEQ   1000
#define BATCH 8192
#define NIN   11

union BfOp { unsigned u[8]; v16bf v; };

// round-to-nearest-even f32 -> bf16, pack two into one dword (lo = a, hi = b)
__device__ __forceinline__ unsigned pack_bf16(float a, float b) {
  unsigned ua = __float_as_uint(a);
  unsigned ub = __float_as_uint(b);
  ua += 0x7fffu + ((ua >> 16) & 1u);
  ub += 0x7fffu + ((ub >> 16) & 1u);
  return (ua >> 16) | (ub & 0xffff0000u);
}

__device__ __forceinline__ float tanh_fast(float x) {
#if __has_builtin(__builtin_amdgcn_tanhf)
  return __builtin_amdgcn_tanhf(x);           // v_tanh_f32 (gfx1250 TRANS op)
#else
  float e = __builtin_amdgcn_exp2f(x * 2.885390081777927f);  // exp(2x)
  return 1.0f - 2.0f / (e + 1.0f);
#endif
}

// D = A(16x32 bf16) x B(32x16 bf16) + 0   (C = inline literal zero)
__device__ __forceinline__ v8f wmma_bf16_z(const BfOp& a, const BfOp& b) {
  v8f z = {0, 0, 0, 0, 0, 0, 0, 0};
  return __builtin_amdgcn_wmma_f32_16x16x32_bf16(false, a.v, false, b.v,
                                                 (short)0, z, false, false);
}

__global__ __launch_bounds__(64) void lstm2_wmma_kernel(
    const float* __restrict__ input,
    const float* __restrict__ Wih1, const float* __restrict__ Whh1,
    const float* __restrict__ bih1, const float* __restrict__ bhh1,
    const float* __restrict__ Wih2, const float* __restrict__ Whh2,
    const float* __restrict__ bih2, const float* __restrict__ bhh2,
    const float* __restrict__ Wm1,  const float* __restrict__ bm1,
    const float* __restrict__ Wm2,  const float* __restrict__ bm2,
    float* __restrict__ out)
{
  const int lane   = threadIdx.x & 31;
  const int waveId = blockIdx.x * 2 + (threadIdx.x >> 5);
  const int B0     = waveId * 16;     // 16-batch tile per wave
  const int nIdx   = lane & 15;       // batch column / A row within tile
  const int hi     = lane >> 4;       // lane half selects K half

  // ---- gather static weights into WMMA A-operand layout (rows = gates) ----
  // k order used here (B matches): lane<16: V0..7 K=0..15 (pairs);
  //                          lane>=16: K=16..31. Row M = lane%16.
  BfOp A1[4], A2[4];
#pragma unroll
  for (int g = 0; g < 4; ++g) {
    const int row = g * 16 + nIdx;    // global gate index (i,f,g,o tiles)
#pragma unroll
    for (int v = 0; v < 8; ++v) {
      const int k0 = 2 * v + 16 * hi;
      const int k1 = k0 + 1;
      // layer 1: K=0..10 -> Wih1, K=11..15 -> 0 pad, K=16..31 -> Whh1
      float a0 = (k0 < 16) ? ((k0 < NIN) ? Wih1[row * NIN + k0] : 0.0f)
                           : Whh1[row * 16 + (k0 - 16)];
      float a1 = (k1 < 16) ? ((k1 < NIN) ? Wih1[row * NIN + k1] : 0.0f)
                           : Whh1[row * 16 + (k1 - 16)];
      A1[g].u[v] = pack_bf16(a0, a1);
      // layer 2: K=0..15 -> Wih2 (input = h1), K=16..31 -> Whh2
      a0 = (k0 < 16) ? Wih2[row * 16 + k0] : Whh2[row * 16 + (k0 - 16)];
      a1 = (k1 < 16) ? Wih2[row * 16 + k1] : Whh2[row * 16 + (k1 - 16)];
      A2[g].u[v] = pack_bf16(a0, a1);
    }
  }

  // ---- per-lane biases, pre-scaled for sigmoid fusion ----
  // sig(x+b) = 0.5*tanh(0.5*x + 0.5*b) + 0.5  -> fma folds bias for free
  float hbI1[8], hbF1[8], bG1[8], hbO1[8];
  float hbI2[8], hbF2[8], bG2[8], hbO2[8];
#pragma unroll
  for (int v = 0; v < 8; ++v) {
    const int r = v + 8 * hi;         // C/D row for VGPR v
    hbI1[v] = 0.5f * (bih1[r]      + bhh1[r]);
    hbF1[v] = 0.5f * (bih1[16 + r] + bhh1[16 + r]);
    bG1[v]  =         bih1[32 + r] + bhh1[32 + r];
    hbO1[v] = 0.5f * (bih1[48 + r] + bhh1[48 + r]);
    hbI2[v] = 0.5f * (bih2[r]      + bhh2[r]);
    hbF2[v] = 0.5f * (bih2[16 + r] + bhh2[16 + r]);
    bG2[v]  =         bih2[32 + r] + bhh2[32 + r];
    hbO2[v] = 0.5f * (bih2[48 + r] + bhh2[48 + r]);
  }

  // ---- recurrent state (registers only) ----
  v8f c1  = {0,0,0,0,0,0,0,0};
  v8f c2v = {0,0,0,0,0,0,0,0};
  unsigned p1[4] = {0,0,0,0}, p2[4] = {0,0,0,0};  // own-half packed h pairs
  unsigned q1[4] = {0,0,0,0}, q2[4] = {0,0,0,0};  // partner-half pairs
  float h2v[8] = {0,0,0,0,0,0,0,0};

  float xr[6];
  auto load_x = [&](int t) {
    // lo lanes load x[b,0..5]; hi lanes load x[b,6..10] (NT: streamed once)
    const float* xb = input + (t * BATCH + B0 + nIdx) * NIN + (hi ? 6 : 0);
#pragma unroll
    for (int j = 0; j < 5; ++j) xr[j] = __builtin_nontemporal_load(xb + j);
    xr[5] = hi ? 0.0f : __builtin_nontemporal_load(xb + 5);
  };

  load_x(0);

#pragma unroll 1
  for (int it = 0; it <= SEQ; ++it) {   // 1000 scan steps + extra step on t=4
    // pack this step's x; swap halves so lo lanes own all 11 features
    unsigned pk0 = pack_bf16(xr[0], xr[1]);   // lo:(x0,x1)   hi:(x6,x7)
    unsigned pk1 = pack_bf16(xr[2], xr[3]);   // lo:(x2,x3)   hi:(x8,x9)
    unsigned pk2 = pack_bf16(xr[4], xr[5]);   // lo:(x4,x5)   hi:(x10,0)
    unsigned xq0 = __shfl_xor(pk0, 16, 32);
    unsigned xq1 = __shfl_xor(pk1, 16, 32);
    unsigned xq2 = __shfl_xor(pk2, 16, 32);

    // prefetch next step's x (hides global latency behind WMMA+VALU)
    if (it < SEQ) load_x(it == SEQ - 1 ? 4 : it + 1);

    // B1 (32x16): lo lanes K=0..15 = [x|pad], hi lanes K=16..31 = h1[0..15]
    BfOp bop1;
    bop1.u[0] = hi ? q1[0] : pk0;
    bop1.u[1] = hi ? q1[1] : pk1;
    bop1.u[2] = hi ? q1[2] : pk2;
    bop1.u[3] = hi ? q1[3] : xq0;
    bop1.u[4] = hi ? p1[0] : xq1;
    bop1.u[5] = hi ? p1[1] : xq2;
    bop1.u[6] = hi ? p1[2] : 0u;
    bop1.u[7] = hi ? p1[3] : 0u;

    v8f di = wmma_bf16_z(A1[0], bop1);
    v8f df = wmma_bf16_z(A1[1], bop1);
    v8f dg = wmma_bf16_z(A1[2], bop1);
    v8f dq = wmma_bf16_z(A1[3], bop1);

    float h1v[8];
#pragma unroll
    for (int v = 0; v < 8; ++v) {
      float si = 0.5f + 0.5f * tanh_fast(fmaf(0.5f, di[v], hbI1[v]));
      float sf = 0.5f + 0.5f * tanh_fast(fmaf(0.5f, df[v], hbF1[v]));
      float tg = tanh_fast(dg[v] + bG1[v]);
      float so = 0.5f + 0.5f * tanh_fast(fmaf(0.5f, dq[v], hbO1[v]));
      float cn = sf * c1[v] + si * tg;
      c1[v] = cn;
      h1v[v] = so * tanh_fast(cn);
    }
#pragma unroll
    for (int j = 0; j < 4; ++j) {
      p1[j] = pack_bf16(h1v[2 * j], h1v[2 * j + 1]);
      q1[j] = __shfl_xor(p1[j], 16, 32);
    }

    // B2: lo lanes K=0..15 = h1[0..15], hi lanes K=16..31 = h2[0..15]
    BfOp bop2;
    bop2.u[0] = hi ? q2[0] : p1[0];
    bop2.u[1] = hi ? q2[1] : p1[1];
    bop2.u[2] = hi ? q2[2] : p1[2];
    bop2.u[3] = hi ? q2[3] : p1[3];
    bop2.u[4] = hi ? p2[0] : q1[0];
    bop2.u[5] = hi ? p2[1] : q1[1];
    bop2.u[6] = hi ? p2[2] : q1[2];
    bop2.u[7] = hi ? p2[3] : q1[3];

    v8f ei = wmma_bf16_z(A2[0], bop2);
    v8f ef = wmma_bf16_z(A2[1], bop2);
    v8f eg = wmma_bf16_z(A2[2], bop2);
    v8f eo = wmma_bf16_z(A2[3], bop2);

#pragma unroll
    for (int v = 0; v < 8; ++v) {
      float si = 0.5f + 0.5f * tanh_fast(fmaf(0.5f, ei[v], hbI2[v]));
      float sf = 0.5f + 0.5f * tanh_fast(fmaf(0.5f, ef[v], hbF2[v]));
      float tg = tanh_fast(eg[v] + bG2[v]);
      float so = 0.5f + 0.5f * tanh_fast(fmaf(0.5f, eo[v], hbO2[v]));
      float cn = sf * c2v[v] + si * tg;
      c2v[v] = cn;
      h2v[v] = so * tanh_fast(cn);
    }
#pragma unroll
    for (int j = 0; j < 4; ++j) {
      p2[j] = pack_bf16(h2v[2 * j], h2v[2 * j + 1]);
      q2[j] = __shfl_xor(p2[j], 16, 32);
    }
  }

  // ---- collapsed linear head: out = wv . h2 + bias0 ----
  float partial = 0.0f;
#pragma unroll
  for (int v = 0; v < 8; ++v) {
    const int h = v + 8 * hi;
    float wv = 0.0f;
#pragma unroll
    for (int j = 0; j < 8; ++j) wv += Wm2[j] * Wm1[j * 16 + h];
    partial += wv * h2v[v];
  }
  float tot = partial + __shfl_xor(partial, 16, 32);
  if (!hi) {
    float bias0 = bm2[0];
#pragma unroll
    for (int j = 0; j < 8; ++j) bias0 += Wm2[j] * bm1[j];
    out[B0 + nIdx] = tot + bias0;
  }
}

extern "C" void kernel_launch(void* const* d_in, const int* in_sizes, int n_in,
                              void* d_out, int out_size, void* d_ws, size_t ws_size,
                              hipStream_t stream) {
  (void)in_sizes; (void)n_in; (void)d_ws; (void)ws_size; (void)out_size;
  const float* input = (const float*)d_in[0];
  const float* Wih1  = (const float*)d_in[1];
  const float* Whh1  = (const float*)d_in[2];
  const float* bih1  = (const float*)d_in[3];
  const float* bhh1  = (const float*)d_in[4];
  const float* Wih2  = (const float*)d_in[5];
  const float* Whh2  = (const float*)d_in[6];
  const float* bih2  = (const float*)d_in[7];
  const float* bhh2  = (const float*)d_in[8];
  const float* Wm1   = (const float*)d_in[9];
  const float* bm1   = (const float*)d_in[10];
  const float* Wm2   = (const float*)d_in[11];
  const float* bm2   = (const float*)d_in[12];

  dim3 grid(BATCH / 32);   // 2 waves/block * 16 batch/wave -> 256 blocks
  dim3 block(64);
  lstm2_wmma_kernel<<<grid, block, 0, stream>>>(
      input, Wih1, Whh1, bih1, bhh1, Wih2, Whh2, bih2, bhh2,
      Wm1, bm1, Wm2, bm2, (float*)d_out);
}
